// SupernodeToBgGraphPropagator_60765197304219
// MI455X (gfx1250) — matrix operand order repaired and checked by reading, not compiled, not run on hardware
//
#include <hip/hip_runtime.h>

#define N_NODES 200000
#define N_SUPER 20000
#define N_EDGES 640000
#define EMB     256

typedef __attribute__((ext_vector_type(16))) __bf16 v16bf;
typedef __attribute__((ext_vector_type(8)))  __bf16 v8bf;
typedef __attribute__((ext_vector_type(8)))  float  v8f;
typedef __attribute__((ext_vector_type(4)))  int    v4i;

#if __has_builtin(__builtin_amdgcn_global_load_async_to_lds_b128) && \
    __has_builtin(__builtin_amdgcn_s_wait_asynccnt)
#define HAVE_ASYNC_LDS 1
#else
#define HAVE_ASYNC_LDS 0
#endif

__device__ __forceinline__ unsigned short f32_to_bf16_raw(float f) {
  union { float f; unsigned u; } v; v.f = f;
  unsigned r = v.u + 0x7FFFu + ((v.u >> 16) & 1u);   // round-to-nearest-even
  return (unsigned short)(r >> 16);
}
__device__ __forceinline__ __bf16 f32_to_bf16(float f) {
  union { unsigned short s; __bf16 b; } o;
  o.s = f32_to_bf16_raw(f);
  return o.b;
}
__device__ __forceinline__ unsigned pack2(float lo, float hi) {
  return (unsigned)f32_to_bf16_raw(lo) | ((unsigned)f32_to_bf16_raw(hi) << 16);
}

// ---------------------------------------------------------------------------
// K0: pre-swizzle W1/W2 into WMMA B-fragment order (B[k][n] = W[n][k], bf16).
// Packed element index: (((kc*16 + nt)*32 + lane)*16 + j)
//   lane: n = nt*16 + (lane&15);  k = kc*32 + (j>>3)*16 + (lane>>4)*8 + (j&7)   (ISA 7.12.2)
// => each lane's 16-element fragment is 32 contiguous bytes.
// ---------------------------------------------------------------------------
__global__ void __launch_bounds__(256) pack_weights(
    const float* __restrict__ W1, const float* __restrict__ W2,
    unsigned short* __restrict__ w1b, unsigned short* __restrict__ w2b) {
  int i = blockIdx.x * blockDim.x + threadIdx.x;      // 0..131071
  const float* W          = (i < 65536) ? W1  : W2;
  unsigned short* dst     = (i < 65536) ? w1b : w2b;
  int idx  = i & 65535;
  int j    = idx & 15;
  int lane = (idx >> 4) & 31;
  int nt   = (idx >> 9) & 15;
  int kc   = (idx >> 13) & 7;
  int n = nt * 16 + (lane & 15);
  int k = kc * 32 + (j >> 3) * 16 + (lane >> 4) * 8 + (j & 7);
  dst[idx] = f32_to_bf16_raw(W[n * EMB + k]);
}

// ---------------------------------------------------------------------------
// K1: out = x (float4 copies) and x1h = bf16(x) (one b128 store per 8 elems).
// ---------------------------------------------------------------------------
__global__ void __launch_bounds__(256) copy_and_cvt(
    const float* __restrict__ x, float* __restrict__ out,
    unsigned short* __restrict__ x1h) {
  long long t = (long long)blockIdx.x * blockDim.x + threadIdx.x;
  long long base = t * 8;
  float4 a = ((const float4*)(x + base))[0];
  float4 b = ((const float4*)(x + base))[1];
  ((float4*)(out + base))[0] = a;
  ((float4*)(out + base))[1] = b;
  uint4 u;
  u.x = pack2(a.x, a.y); u.y = pack2(a.z, a.w);
  u.z = pack2(b.x, b.y); u.w = pack2(b.z, b.w);
  *(uint4*)(x1h + base) = u;
}

// ---------------------------------------------------------------------------
// K2: sn_proj = new_supernode_x @ W1^T + b1 ; out[sn_idx] += sn_proj ;
//     refresh bf16 mirror x1h for those rows (step-2 reads post-update x1).
// One block = 16 supernode rows x 256 cols; 8 waves, 2 N-tiles each.
// ---------------------------------------------------------------------------
__global__ void __launch_bounds__(256) supernode_gemm(
    const float* __restrict__ snx, const unsigned short* __restrict__ w1b,
    const float* __restrict__ b1, const int* __restrict__ sn_idx,
    float* __restrict__ out, unsigned short* __restrict__ x1h) {
  const int tile = blockIdx.x;            // 0..1249 (20000/16 exact)
  const int lane = threadIdx.x & 31;
  const int wave = threadIdx.x >> 5;      // 0..7
  const int h    = lane >> 4;
  const int m    = lane & 15;
  const float* arow = snx + (long long)(tile * 16 + m) * EMB;
  v8f acc0 = {}; v8f acc1 = {};
#pragma unroll
  for (int kc = 0; kc < 8; ++kc) {
    const int kb = kc * 32;
    // A fragment (16x32 bf16): el[0..7] K=kb+h*8+j ; el[8..15] K=kb+16+h*8+j
    float4 f0 = ((const float4*)(arow + kb + h * 8))[0];
    float4 f1 = ((const float4*)(arow + kb + h * 8))[1];
    float4 f2 = ((const float4*)(arow + kb + 16 + h * 8))[0];
    float4 f3 = ((const float4*)(arow + kb + 16 + h * 8))[1];
    v16bf a;
    a[0]=f32_to_bf16(f0.x);  a[1]=f32_to_bf16(f0.y);  a[2]=f32_to_bf16(f0.z);  a[3]=f32_to_bf16(f0.w);
    a[4]=f32_to_bf16(f1.x);  a[5]=f32_to_bf16(f1.y);  a[6]=f32_to_bf16(f1.z);  a[7]=f32_to_bf16(f1.w);
    a[8]=f32_to_bf16(f2.x);  a[9]=f32_to_bf16(f2.y);  a[10]=f32_to_bf16(f2.z); a[11]=f32_to_bf16(f2.w);
    a[12]=f32_to_bf16(f3.x); a[13]=f32_to_bf16(f3.y); a[14]=f32_to_bf16(f3.z); a[15]=f32_to_bf16(f3.w);
    v16bf bv0 = *(const v16bf*)(w1b + (((kc * 16 + wave * 2    ) * 32 + lane) << 4));
    v16bf bv1 = *(const v16bf*)(w1b + (((kc * 16 + wave * 2 + 1) * 32 + lane) << 4));
    acc0 = __builtin_amdgcn_wmma_f32_16x16x32_bf16(false, a, false, bv0, (short)0, acc0, false, false);
    acc1 = __builtin_amdgcn_wmma_f32_16x16x32_bf16(false, a, false, bv1, (short)0, acc1, false, false);
  }
  const int n0 = wave * 32;
  const float bias0 = b1[n0 + m];
  const float bias1 = b1[n0 + 16 + m];
#pragma unroll
  for (int v = 0; v < 8; ++v) {
    int grow = sn_idx[tile * 16 + v + 8 * h];          // unique rows -> no atomics
    long long o0 = (long long)grow * EMB + n0 + m;
    float nv0 = out[o0]      + acc0[v] + bias0;
    float nv1 = out[o0 + 16] + acc1[v] + bias1;
    out[o0]      = nv0;  x1h[o0]      = f32_to_bf16_raw(nv0);
    out[o0 + 16] = nv1;  x1h[o0 + 16] = f32_to_bf16_raw(nv1);
  }
}

// ---------------------------------------------------------------------------
// K3: msg = x1[src] @ W2^T + b2 ; atomic scatter-add into out[dst].
// One block = 16 edges; gather 16 bf16 rows (512B each) into LDS, padded
// row stride 264 shorts (528B -> 4-dword bank skew, conflict-free b128 reads).
// Staging uses GLOBAL_LOAD_ASYNC_TO_LDS_B128 (ASYNCcnt) when available.
// ---------------------------------------------------------------------------
__global__ void __launch_bounds__(256) edge_gemm(
    const unsigned short* __restrict__ x1h, const unsigned short* __restrict__ w2b,
    const float* __restrict__ b2, const int* __restrict__ edge,  // [2*E]: dst then src
    float* __restrict__ out) {
  __shared__ unsigned short As[16 * 264];
  const int tile = blockIdx.x;            // 0..39999 (640000/16 exact)
  const int t = threadIdx.x;
  {
    int r = t >> 4;                       // row 0..15
    int c = t & 15;                       // 32B chunk 0..15
    int srcRow = edge[N_EDGES + tile * 16 + r];
    const unsigned short* g = x1h + (long long)srcRow * EMB + c * 16;
    unsigned short* s = &As[r * 264 + c * 16];
#if HAVE_ASYNC_LDS
    // direct global->LDS DMA path, tracked by ASYNCcnt (no VGPR round-trip)
    unsigned short* gm = (unsigned short*)g;   // drop const
    auto gp = (__attribute__((address_space(1))) v4i*)gm;
    auto lp = (__attribute__((address_space(3))) v4i*)s;
    __builtin_amdgcn_global_load_async_to_lds_b128(gp,     lp,     0, 0);
    __builtin_amdgcn_global_load_async_to_lds_b128(gp + 1, lp + 1, 0, 0);
    __builtin_amdgcn_s_wait_asynccnt(0);
#else
    uint4 d0 = ((const uint4*)g)[0], d1 = ((const uint4*)g)[1];
    ((uint4*)s)[0] = d0; ((uint4*)s)[1] = d1;
#endif
  }
  __syncthreads();
  const int lane = t & 31;
  const int wave = t >> 5;
  const int h = lane >> 4;
  const int m = lane & 15;
  v8f acc0 = {}; v8f acc1 = {};
#pragma unroll
  for (int kc = 0; kc < 8; ++kc) {
    const int kb = kc * 32;
    v8bf lo = *(const v8bf*)(&As[m * 264 + kb + h * 8]);        // ds_load_b128
    v8bf hi = *(const v8bf*)(&As[m * 264 + kb + 16 + h * 8]);   // ds_load_b128
    v16bf a = __builtin_shufflevector(lo, hi, 0,1,2,3,4,5,6,7,8,9,10,11,12,13,14,15);
    v16bf bv0 = *(const v16bf*)(w2b + (((kc * 16 + wave * 2    ) * 32 + lane) << 4));
    v16bf bv1 = *(const v16bf*)(w2b + (((kc * 16 + wave * 2 + 1) * 32 + lane) << 4));
    acc0 = __builtin_amdgcn_wmma_f32_16x16x32_bf16(false, a, false, bv0, (short)0, acc0, false, false);
    acc1 = __builtin_amdgcn_wmma_f32_16x16x32_bf16(false, a, false, bv1, (short)0, acc1, false, false);
  }
  const int n0 = wave * 32;
  const float bias0 = b2[n0 + m];
  const float bias1 = b2[n0 + 16 + m];
#pragma unroll
  for (int v = 0; v < 8; ++v) {
    int drow = edge[tile * 16 + v + 8 * h];
    long long o0 = (long long)drow * EMB + n0 + m;
    atomicAdd(&out[o0],      acc0[v] + bias0);
    atomicAdd(&out[o0 + 16], acc1[v] + bias1);
  }
}

// ---------------------------------------------------------------------------
extern "C" void kernel_launch(void* const* d_in, const int* in_sizes, int n_in,
                              void* d_out, int out_size, void* d_ws, size_t ws_size,
                              hipStream_t stream) {
  const float* x    = (const float*)d_in[0];
  const float* snx  = (const float*)d_in[1];
  const int*   edge = (const int*)d_in[2];   // [2, E] flat: dst row then src row
  const int*   sidx = (const int*)d_in[3];
  // d_in[4] graph_batch: unused by forward
  const float* W1   = (const float*)d_in[5];
  const float* b1   = (const float*)d_in[6];
  const float* W2   = (const float*)d_in[7];
  const float* b2   = (const float*)d_in[8];
  float* out = (float*)d_out;

  // workspace: [W1 packed 128KB][W2 packed 128KB][x1h bf16 102.4MB] ~= 102.7MB
  unsigned short* w1b = (unsigned short*)d_ws;
  unsigned short* w2b = w1b + 65536;
  unsigned short* x1h = w2b + 65536;

  pack_weights <<<512, 256, 0, stream>>>(W1, W2, w1b, w2b);
  copy_and_cvt <<<(N_NODES * (EMB / 8)) / 256, 256, 0, stream>>>(x, out, x1h);
  supernode_gemm<<<N_SUPER / 16, 256, 0, stream>>>(snx, w1b, b1, sidx, out, x1h);
  edge_gemm    <<<N_EDGES / 16, 256, 0, stream>>>(x1h, w2b, b2, edge, out);
}
